// GroupedQueryAttention_37280316129392
// MI455X (gfx1250) — matrix operand order repaired and checked by reading, not compiled, not run on hardware
//
#include <hip/hip_runtime.h>
#include <hip/hip_bf16.h>

typedef _Float16 v16h __attribute__((ext_vector_type(16)));
typedef _Float16 v8h  __attribute__((ext_vector_type(8)));
typedef _Float16 v4h  __attribute__((ext_vector_type(4)));
typedef float    v8f  __attribute__((ext_vector_type(8)));

#define S_LEN   2048
#define D_MODEL 2048
#define N_HQ    32
#define N_HKV   8
#define DK_SZ   64
#define G_SZ    4

union AFrag { v16h v; v8h h[2]; };

// ---------------------------------------------------------------------------
// f32 -> f16 conversion (vectorized x4)
// ---------------------------------------------------------------------------
__global__ __launch_bounds__(256) void cvt_f32_f16(const float* __restrict__ in,
                                                   _Float16* __restrict__ out, int n) {
    int i = (blockIdx.x * blockDim.x + threadIdx.x) * 4;
    if (i < n) {
        float4 f = *(const float4*)(in + i);
        v4h h = { (_Float16)f.x, (_Float16)f.y, (_Float16)f.z, (_Float16)f.w };
        *(v4h*)(out + i) = h;
    }
}

// ---------------------------------------------------------------------------
// Generic GEMM: Y[M,N] = scale * (X[M,K] @ W[N,K]^T)   (torch Linear form)
// X,W f16 row-major. Block = 256 thr = 8 waves (2 M x 4 N); block tile 32x256,
// wave tile 16x64 (4 WMMA accumulators), K-step 32 per v_wmma_f32_16x16x32_f16.
// MODE 0: f16 row-major out.  MODE 1: f16 transposed out (ld=ldT).  MODE 2: f32 out.
// ---------------------------------------------------------------------------
template<int MODE>
__global__ __launch_bounds__(256, 1)
void gemm_xwT(const _Float16* __restrict__ X, const _Float16* __restrict__ W,
              void* __restrict__ Y, int M, int N, int K, int ldT, float scale) {
    const int lane  = threadIdx.x & 31;
    const int wid   = threadIdx.x >> 5;
    const int l16   = lane & 15;
    const int lhalf = lane >> 4;
    const int mBase = blockIdx.y * 32 + (wid >> 2) * 16;
    const int nBase = blockIdx.x * 256 + (wid & 3) * 64;

    const _Float16* xrow = X + (size_t)(mBase + l16) * K;
    v8f acc[4] = {};

    for (int k = 0; k < K; k += 32) {
        if (k + 256 < K) __builtin_prefetch(xrow + k + 256, 0, 1);
        AFrag a;
        a.h[0] = *(const v8h*)(xrow + k + lhalf * 16);
        a.h[1] = *(const v8h*)(xrow + k + lhalf * 16 + 8);
#pragma unroll
        for (int nt = 0; nt < 4; ++nt) {
            const _Float16* wp = W + (size_t)(nBase + nt * 16 + l16) * K + k + lhalf * 16;
            v16h b = *(const v16h*)wp;
            acc[nt] = __builtin_amdgcn_wmma_f32_16x16x32_f16(
                false, a.v, false, b, (short)0, acc[nt], false, false);
        }
    }

#pragma unroll
    for (int nt = 0; nt < 4; ++nt) {
        int col = nBase + nt * 16 + l16;
#pragma unroll
        for (int r = 0; r < 8; ++r) {
            int row = mBase + r + lhalf * 8;
            float v = acc[nt][r] * scale;
            if constexpr (MODE == 0)
                ((_Float16*)Y)[(size_t)row * N + col] = (_Float16)v;
            else if constexpr (MODE == 1)
                ((_Float16*)Y)[(size_t)col * ldT + row] = (_Float16)v;
            else
                ((float*)Y)[(size_t)row * N + col] = v;
        }
    }
}

// ---------------------------------------------------------------------------
// Fused GQA attention for one (b, hq, 16-query-row tile).
// Block = 512 thr = 16 waves; wave w owns key columns [w*128, w*128+128),
// i.e. 8 WMMA score tiles per wave (64 acc VGPRs -> no scratch spills).
// Phase 1: scores (Q pre-scaled by 1/8) via WMMA.
// Phase 2: tile-resident softmax (shuffle + LDS reductions), single-pass
//          normalized f32 write to attn output, f16 copy into LDS P-buffer.
// Phase 3: P @ V (V pre-transposed) via WMMA, cross-wave LDS reduction,
//          f16 context written for the output projection.
// ---------------------------------------------------------------------------
__global__ __launch_bounds__(512, 1)
void gqa_attn(const _Float16* __restrict__ Qp,   // [B*S, 2048] (scaled)
              const _Float16* __restrict__ Kp,   // [B*S, 512]
              const _Float16* __restrict__ Vt,   // [512, B*S] transposed
              float* __restrict__ attnOut,       // [B, HQ, S, S]
              _Float16* __restrict__ Ctx,        // [B*S, 2048]
              int Mtot) {
    extern __shared__ __align__(16) char smem[];
    _Float16* Pbuf = (_Float16*)smem;                 // [16][2048] f16 = 64KB
    float*    red  = (float*)(smem + 16 * 2048 * 2);  // [16][16]
    float*    rowM = red + 16 * 16;                   // [16]
    float*    rowS = rowM + 16;                       // [16]

    const int b     = blockIdx.x / N_HQ;
    const int hq    = blockIdx.x % N_HQ;
    const int hkv   = hq / G_SZ;
    const int qBase = blockIdx.y * 16;
    const int lane  = threadIdx.x & 31;
    const int wid   = threadIdx.x >> 5;
    const int l16   = lane & 15;
    const int lhalf = lane >> 4;
    const int jWave = wid * 128;

    // ---- Q fragments (16 rows x 64 dk, two K-steps of 32) ----
    const _Float16* qrow = Qp + (size_t)(b * S_LEN + qBase + l16) * D_MODEL + hq * DK_SZ;
    AFrag aq[2];
#pragma unroll
    for (int kk = 0; kk < 2; ++kk) {
        aq[kk].h[0] = *(const v8h*)(qrow + kk * 32 + lhalf * 16);
        aq[kk].h[1] = *(const v8h*)(qrow + kk * 32 + lhalf * 16 + 8);
    }

    // ---- Phase 1: scores for 16 x 128 (per wave), 8 WMMA tiles ----
    v8f acc[8];
#pragma unroll
    for (int nt = 0; nt < 8; ++nt) {
        int j0 = jWave + nt * 16;
        v8f c = {};
#pragma unroll
        for (int kk = 0; kk < 2; ++kk) {
            const _Float16* kp = Kp + (size_t)(b * S_LEN + j0 + l16) * (N_HKV * DK_SZ)
                                 + hkv * DK_SZ + kk * 32 + lhalf * 16;
            v16h bfr = *(const v16h*)kp;
            c = __builtin_amdgcn_wmma_f32_16x16x32_f16(
                false, aq[kk].v, false, bfr, (short)0, c, false, false);
        }
        acc[nt] = c;
    }

    // ---- Phase 2a: row max ----
    float pm[8];
#pragma unroll
    for (int r = 0; r < 8; ++r) pm[r] = -3.0e38f;
#pragma unroll
    for (int nt = 0; nt < 8; ++nt)
#pragma unroll
        for (int r = 0; r < 8; ++r) pm[r] = fmaxf(pm[r], acc[nt][r]);
#pragma unroll
    for (int m = 1; m < 16; m <<= 1)
#pragma unroll
        for (int r = 0; r < 8; ++r) pm[r] = fmaxf(pm[r], __shfl_xor(pm[r], m, 32));
    if (l16 == 0)
#pragma unroll
        for (int r = 0; r < 8; ++r) red[wid * 16 + lhalf * 8 + r] = pm[r];
    __syncthreads();
    if (threadIdx.x < 16) {
        float m = -3.0e38f;
        for (int w2 = 0; w2 < 16; ++w2) m = fmaxf(m, red[w2 * 16 + threadIdx.x]);
        rowM[threadIdx.x] = m;
    }
    __syncthreads();
    float rmax[8];
#pragma unroll
    for (int r = 0; r < 8; ++r) rmax[r] = rowM[lhalf * 8 + r];

    // ---- Phase 2b: exp + row sum (overwrite acc with exp values) ----
    float ps[8] = {0.f, 0.f, 0.f, 0.f, 0.f, 0.f, 0.f, 0.f};
#pragma unroll
    for (int nt = 0; nt < 8; ++nt)
#pragma unroll
        for (int r = 0; r < 8; ++r) {
            float e = __expf(acc[nt][r] - rmax[r]);
            acc[nt][r] = e;
            ps[r] += e;
        }
#pragma unroll
    for (int m = 1; m < 16; m <<= 1)
#pragma unroll
        for (int r = 0; r < 8; ++r) ps[r] += __shfl_xor(ps[r], m, 32);
    __syncthreads();  // red reuse
    if (l16 == 0)
#pragma unroll
        for (int r = 0; r < 8; ++r) red[wid * 16 + lhalf * 8 + r] = ps[r];
    __syncthreads();
    if (threadIdx.x < 16) {
        float s = 0.f;
        for (int w2 = 0; w2 < 16; ++w2) s += red[w2 * 16 + threadIdx.x];
        rowS[threadIdx.x] = s;
    }
    __syncthreads();
    float rinv[8];
#pragma unroll
    for (int r = 0; r < 8; ++r) rinv[r] = 1.0f / rowS[lhalf * 8 + r];

    // ---- Phase 2c: single-pass normalized write (f32 out + f16 LDS) ----
    float* attnBase = attnOut + ((size_t)(b * N_HQ + hq) * S_LEN + qBase) * S_LEN;
#pragma unroll
    for (int nt = 0; nt < 8; ++nt) {
        int col = jWave + nt * 16 + l16;
#pragma unroll
        for (int r = 0; r < 8; ++r) {
            int row = r + lhalf * 8;
            float p = acc[nt][r] * rinv[r];
            attnBase[(size_t)row * S_LEN + col] = p;
            Pbuf[row * 2048 + col] = (_Float16)p;
        }
    }
    __syncthreads();

    // ---- Phase 3: ctx partial = P(16 x 128) @ V(128 x 64) per wave ----
    v8f ctx[4] = {};
#pragma unroll
    for (int step = 0; step < 4; ++step) {
        int j0 = jWave + step * 32;
        AFrag ap;
        const _Float16* pb = Pbuf + l16 * 2048 + j0 + lhalf * 16;
        ap.h[0] = *(const v8h*)(pb);
        ap.h[1] = *(const v8h*)(pb + 8);
#pragma unroll
        for (int nt2 = 0; nt2 < 4; ++nt2) {
            const _Float16* vp = Vt + (size_t)(hkv * DK_SZ + nt2 * 16 + l16) * Mtot
                                 + b * S_LEN + j0 + lhalf * 16;
            v16h bfr = *(const v16h*)vp;
            ctx[nt2] = __builtin_amdgcn_wmma_f32_16x16x32_f16(
                false, ap.v, false, bfr, (short)0, ctx[nt2], false, false);
        }
    }
    __syncthreads();  // all waves done reading Pbuf before alias overwrite

    // ---- cross-wave reduction of 16x64 ctx via LDS (aliases Pbuf, 64KB) ----
    float* cred = (float*)smem;  // [16][16][64]
#pragma unroll
    for (int nt2 = 0; nt2 < 4; ++nt2) {
        int col = nt2 * 16 + l16;
#pragma unroll
        for (int r = 0; r < 8; ++r) {
            int row = r + lhalf * 8;
            cred[((size_t)wid * 16 + row) * 64 + col] = ctx[nt2][r];
        }
    }
    __syncthreads();
    for (int e = threadIdx.x; e < 16 * 64; e += 512) {
        int row = e >> 6, col = e & 63;
        float s = 0.f;
        for (int w2 = 0; w2 < 16; ++w2) s += cred[((size_t)w2 * 16 + row) * 64 + col];
        Ctx[(size_t)(b * S_LEN + qBase + row) * D_MODEL + hq * DK_SZ + col] = (_Float16)s;
    }
}

// ---------------------------------------------------------------------------
extern "C" void kernel_launch(void* const* d_in, const int* in_sizes, int n_in,
                              void* d_out, int out_size, void* d_ws, size_t ws_size,
                              hipStream_t stream) {
    (void)n_in; (void)out_size; (void)ws_size;
    const float* q_f32  = (const float*)d_in[0];
    const float* k_f32  = (const float*)d_in[1];
    const float* v_f32  = (const float*)d_in[2];
    const float* wq_f32 = (const float*)d_in[3];
    const float* wk_f32 = (const float*)d_in[4];
    const float* wv_f32 = (const float*)d_in[5];
    const float* wo_f32 = (const float*)d_in[6];

    const int S = S_LEN, DM = D_MODEL;
    const int B = in_sizes[0] / (S * DM);       // = 2
    const int M = B * S;                        // = 4096
    const int NKV = N_HKV * DK_SZ;              // = 512

    // workspace layout (f16 elements)
    _Float16* w = (_Float16*)d_ws;
    size_t off = 0;
    _Float16* q16  = w + off; off += (size_t)M * DM;
    _Float16* k16  = w + off; off += (size_t)M * DM;
    _Float16* v16  = w + off; off += (size_t)M * DM;
    _Float16* wq16 = w + off; off += (size_t)DM * DM;
    _Float16* wk16 = w + off; off += (size_t)NKV * DM;
    _Float16* wv16 = w + off; off += (size_t)NKV * DM;
    _Float16* wo16 = w + off; off += (size_t)DM * DM;
    _Float16* Qp   = w + off; off += (size_t)M * DM;
    _Float16* Kp   = w + off; off += (size_t)M * NKV;
    _Float16* Vt   = w + off; off += (size_t)M * NKV;
    _Float16* Ctx  = w + off; off += (size_t)M * DM;

    float* out  = (float*)d_out;                 // [B,S,DM]
    float* attn = out + (size_t)B * S * DM;      // [B,HQ,S,S]

    // 1) convert inputs/weights to f16
    auto cvt = [&](const float* src, _Float16* dst, size_t n) {
        int blocks = (int)((n / 4 + 255) / 256);
        hipLaunchKernelGGL(cvt_f32_f16, dim3(blocks), dim3(256), 0, stream, src, dst, (int)n);
    };
    cvt(q_f32,  q16,  (size_t)M * DM);
    cvt(k_f32,  k16,  (size_t)M * DM);
    cvt(v_f32,  v16,  (size_t)M * DM);
    cvt(wq_f32, wq16, (size_t)DM * DM);
    cvt(wk_f32, wk16, (size_t)NKV * DM);
    cvt(wv_f32, wv16, (size_t)NKV * DM);
    cvt(wo_f32, wo16, (size_t)DM * DM);

    // 2) projections (Q carries the 1/sqrt(DK)=0.125 softmax scale)
    hipLaunchKernelGGL((gemm_xwT<0>), dim3(DM / 256, M / 32), dim3(256), 0, stream,
                       q16, wq16, (void*)Qp, M, DM, DM, 0, 0.125f);
    hipLaunchKernelGGL((gemm_xwT<0>), dim3(NKV / 256, M / 32), dim3(256), 0, stream,
                       k16, wk16, (void*)Kp, M, NKV, DM, 0, 1.0f);
    hipLaunchKernelGGL((gemm_xwT<1>), dim3(NKV / 256, M / 32), dim3(256), 0, stream,
                       v16, wv16, (void*)Vt, M, NKV, DM, M, 1.0f);

    // 3) fused attention (66 KB dynamic LDS: 64KB P-tile + reduction scratch)
    size_t smemBytes = (size_t)16 * 2048 * 2 + (16 * 16 + 32) * sizeof(float);
    hipLaunchKernelGGL(gqa_attn, dim3(B * N_HQ, S / 16), dim3(512), smemBytes, stream,
                       Qp, Kp, Vt, attn, Ctx, M);

    // 4) output projection -> f32
    hipLaunchKernelGGL((gemm_xwT<2>), dim3(DM / 256, M / 32), dim3(256), 0, stream,
                       Ctx, wo16, (void*)out, M, DM, DM, 0, 1.0f);
}
